// MultiboxLoss_80307298500651
// MI455X (gfx1250) — compile-verified, matrix operand order checked by reading, not run on hardware
//
#include <hip/hip_runtime.h>
#include <math.h>

#define BB 64
#define PP 8732
#define CC 81
#define GG 50
#define THRESH 0.5f

typedef __attribute__((ext_vector_type(16))) _Float16 v16h;
typedef __attribute__((ext_vector_type(8)))  float    v8f;

__device__ __forceinline__ float blockReduceSum256(float v, float* sh) {
    int tid = threadIdx.x;
    sh[tid] = v;
    __syncthreads();
    for (int s = 128; s > 0; s >>= 1) {
        if (tid < s) sh[tid] += sh[tid + s];
        __syncthreads();
    }
    float r = sh[0];
    __syncthreads();
    return r;
}

// ---------------- K0: init per-GT argmax keys ----------------
__global__ void k_init(unsigned long long* bestKey) {
    int i = blockIdx.x * 256 + threadIdx.x;
    if (i < BB * GG) bestKey[i] = 0ull;
}

// ---------------- K1: IoU, per-prior best gt, per-gt best prior ----------------
__global__ void k_match(const float* __restrict__ priors,
                        const float* __restrict__ truth,
                        unsigned long long* __restrict__ bestKey,
                        float* __restrict__ ov, int* __restrict__ gi) {
    int b = blockIdx.y;
    int p = blockIdx.x * 256 + threadIdx.x;
    bool valid = p < PP;
    int pc = valid ? p : PP - 1;
    float cx = priors[pc * 4 + 0], cy = priors[pc * 4 + 1];
    float w  = priors[pc * 4 + 2], h  = priors[pc * 4 + 3];
    float px1 = cx - 0.5f * w, py1 = cy - 0.5f * h;
    float px2 = cx + 0.5f * w, py2 = cy + 0.5f * h;
    float ap = (px2 - px1) * (py2 - py1);
    float bestI = -1.0f; int bestG = 0;
    int lane = threadIdx.x & 31;
    for (int g = 0; g < GG; ++g) {
        const float* t = truth + ((size_t)b * GG + g) * 5;
        float t0 = t[0], t1 = t[1], t2 = t[2], t3 = t[3];
        float ag = (t2 - t0) * (t3 - t1);
        float ix = fminf(px2, t2) - fmaxf(px1, t0); ix = fmaxf(ix, 0.0f);
        float iy = fminf(py2, t3) - fmaxf(py1, t1); iy = fmaxf(iy, 0.0f);
        float inter = ix * iy;
        float iou = inter / (ap + ag - inter);
        if (iou > bestI) { bestI = iou; bestG = g; }  // first occurrence wins (argmax)
        // key: higher IoU wins; tie -> lower p wins (matches argmax over axis 0)
        unsigned lo = 0xFFFFFFFFu - (unsigned)p;
        unsigned hiw = __float_as_uint(iou);
        unsigned long long key = valid
            ? (((unsigned long long)hiw << 32) | (unsigned long long)lo) : 0ull;
        // wave32 max-reduce (two 32-bit shuffles per step)
        for (int off = 16; off > 0; off >>= 1) {
            unsigned klo = (unsigned)key, khi = (unsigned)(key >> 32);
            unsigned olo = __shfl_down(klo, off, 32);
            unsigned ohi = __shfl_down(khi, off, 32);
            unsigned long long o = ((unsigned long long)ohi << 32) | olo;
            if (o > key) key = o;
        }
        if (lane == 0) atomicMax(&bestKey[b * GG + g], key);
    }
    if (valid) {
        ov[(size_t)b * PP + p] = bestI;
        gi[(size_t)b * PP + p] = bestG;
    }
}

// ---------------- K2: forced matches, conf_t, pos, smooth-L1 loc loss ----------------
__global__ void k_finalize_match(const float* __restrict__ priors,
                                 const float* __restrict__ truth,
                                 const float* __restrict__ pred_loc,
                                 const unsigned long long* __restrict__ bestKey,
                                 const float* __restrict__ ov_in,
                                 int* __restrict__ gi_conf,   // in: gt idx, out: conf_t
                                 int* __restrict__ posm,
                                 float* __restrict__ locp, int* __restrict__ numpos) {
    __shared__ float sh[256];
    int b = blockIdx.x, tid = threadIdx.x;
    float locsum = 0.0f, cnt = 0.0f;
    for (int p = tid; p < PP; p += 256) {
        size_t bp = (size_t)b * PP + p;
        float ov = ov_in[bp];
        int gidx = gi_conf[bp];
        for (int g = 0; g < GG; ++g) {           // ascending g: last write wins
            unsigned long long k = bestKey[b * GG + g];
            unsigned bpri = 0xFFFFFFFFu - (unsigned)(k & 0xFFFFFFFFull);
            if (bpri == (unsigned)p) { ov = 2.0f; gidx = g; }
        }
        const float* t = truth + ((size_t)b * GG + gidx) * 5;
        float label = t[4];
        int conf = (ov < THRESH) ? 0 : (int)(label + 1.0f);
        int pos = conf > 0;
        gi_conf[bp] = conf;
        posm[bp] = pos;
        if (pos) {
            float m0 = t[0], m1 = t[1], m2 = t[2], m3 = t[3];
            float cx = priors[p * 4], cy = priors[p * 4 + 1];
            float w  = priors[p * 4 + 2], h = priors[p * 4 + 3];
            float tt0 = ((m0 + m2) * 0.5f - cx) / w;
            float tt1 = ((m1 + m3) * 0.5f - cy) / h;
            float tt2 = logf((m2 - m0) / w);
            float tt3 = logf((m3 - m1) / h);
            float tt[4] = { tt0, tt1, tt2, tt3 };
            const float* pl = pred_loc + bp * 4;
            float s = 0.0f;
            #pragma unroll
            for (int d = 0; d < 4; ++d) {
                float ad = fabsf(pl[d] - tt[d]);
                s += (ad < 1.0f) ? 0.5f * ad * ad : ad - 0.5f;
            }
            locsum += s; cnt += 1.0f;
        }
    }
    float ls = blockReduceSum256(locsum, sh);
    float cs = blockReduceSum256(cnt, sh);
    if (tid == 0) { locp[b] = ls; numpos[b] = (int)cs; }
}

// ---------------- K3: single-pass WMMA logsumexp -> ce, loss_mine, sum(pos*ce) ----------------
// Wave handles 16 priors x 81 classes, values held in registers (one HBM pass).
// A = exp(x-max) tile (f16), B = ones: D row-sums are K-permutation invariant,
// so only the documented lane->row mapping (lanes L, L+16 -> row M=L) matters.
__global__ void k_ce_wmma(const float* __restrict__ conf,
                          const int* __restrict__ conf_t,
                          const int* __restrict__ posm,
                          float* __restrict__ cearr, float* __restrict__ lmarr,
                          float* __restrict__ cepos_partial) {
    __shared__ float sh[256];
    int b = blockIdx.y, tid = threadIdx.x;
    int wave = tid >> 5, lane = tid & 31;
    int m = lane & 15, hi = lane >> 4;
    int pbase = blockIdx.x * 128 + wave * 16;
    float contrib = 0.0f;
    if (pbase < PP) {                            // wave-uniform: EXEC stays all-ones
        int p = pbase + m;
        int pc = p < PP ? p : PP - 1;
        const float* row = conf + ((size_t)b * PP + pc) * CC;
        __builtin_prefetch(row + (size_t)128 * CC, 0, 1);   // global_prefetch_b8
        // one pass: load tile to registers, track max (K-tail -> -inf so exp -> 0)
        float vals[48];
        float mx = -__builtin_inff();
        #pragma unroll
        for (int wdw = 0; wdw < 3; ++wdw) {
            #pragma unroll
            for (int hh = 0; hh < 16; ++hh) {
                int K = hh + (hh >= 8 ? 8 : 0) + hi * 8;
                int cls = 32 * wdw + K;
                int clc = cls < CC ? cls : CC - 1;
                float x = row[clc];
                x = (cls < CC) ? x : -__builtin_inff();
                vals[wdw * 16 + hh] = x;
                mx = fmaxf(mx, x);
            }
        }
        float rowmax = fmaxf(mx, __shfl_xor(mx, 16, 32));
        // exp + WMMA row-sum (B = ones), 3 back-to-back wmma issues
        v16h bones;
        #pragma unroll
        for (int hh = 0; hh < 16; ++hh) bones[hh] = (_Float16)1.0f;
        v8f c = {0.f,0.f,0.f,0.f,0.f,0.f,0.f,0.f};
        #pragma unroll
        for (int wdw = 0; wdw < 3; ++wdw) {
            v16h a;
            #pragma unroll
            for (int hh = 0; hh < 16; ++hh) {
                float e = expf(vals[wdw * 16 + hh] - rowmax);   // exp(-inf)=0
                a[hh] = (_Float16)e;
            }
            c = __builtin_amdgcn_wmma_f32_16x16x32_f16(
                    false, a, false, bones, (short)0, c, false, false);
        }
        // D layout: lanes 0-15 vgpr r -> M=r; lanes 16-31 vgpr r -> M=r+8
        float t = c[lane & 7];
        int srcLane = (m < 8) ? m : (m + 8);
        float s = __shfl(t, srcLane, 32);
        if (lane < 16 && p < PP) {
            size_t bp = (size_t)b * PP + p;
            int tc = conf_t[bp];
            float tgt = row[tc];
            float ce = rowmax + logf(s) - tgt;
            int pos = posm[bp];
            cearr[bp] = ce;
            lmarr[bp] = pos ? 0.0f : ce;
            contrib = pos ? ce : 0.0f;
        }
    }
    float tot = blockReduceSum256(contrib, sh);
    if (tid == 0) cepos_partial[blockIdx.y * gridDim.x + blockIdx.x] = tot;
}

// ---------------- K4: hard negative mining via stable ranks ----------------
// neg[j] = (argsort(lm)[j] < num_neg)  <=>  selected j = { rank(i) : i < num_neg }
// rank(i) = #{lm[j] < lm[i]} + #{j<i : lm[j]==lm[i]}   (stable sort)
// Row staged into LDS with async global->LDS b128 DMA (ASYNCcnt-tracked).
__global__ void k_negmine(const float* __restrict__ lmarr,
                          const float* __restrict__ cearr,
                          const int* __restrict__ posm,
                          const int* __restrict__ numpos,
                          float* __restrict__ negce) {
    __shared__ __align__(16) float slm[PP];
    __shared__ float sh[256];
    int b = blockIdx.x, tid = threadIdx.x;
    const float* src = lmarr + (size_t)b * PP;     // 16B-aligned (PP*4 % 16 == 0)
    // PP/4 == 2183 exact: copy row as 16-byte async transfers
    for (int idx = tid; idx < PP / 4; idx += 256) {
        unsigned ldsoff = (unsigned)(unsigned long long)(const void*)&slm[4 * idx];
        const float* g = src + 4 * idx;
        asm volatile("global_load_async_to_lds_b128 %0, %1, off"
                     :: "v"(ldsoff), "v"(g) : "memory");
    }
    asm volatile("s_wait_asynccnt 0x0" ::: "memory");
    __syncthreads();
    int np = numpos[b];
    int nn = 3 * np; if (nn > PP - 1) nn = PP - 1;
    float acc = 0.0f;
    for (int i = tid; i < nn; i += 256) {
        float v = slm[i];
        int cl = 0, ceq = 0;
        for (int j = 0; j < PP; ++j) {
            float lj = slm[j];
            cl  += (lj < v) ? 1 : 0;
            ceq += ((lj == v) && (j < i)) ? 1 : 0;
        }
        int r = cl + ceq;
        size_t br = (size_t)b * PP + r;
        if (!posm[br]) acc += cearr[br];
    }
    float tot = blockReduceSum256(acc, sh);
    if (tid == 0) negce[b] = tot;
}

// ---------------- K5: finalize (serial, deterministic order) ----------------
__global__ void k_final(const float* __restrict__ locp, const int* __restrict__ numpos,
                        const float* __restrict__ negce, const float* __restrict__ cepos,
                        int nCepos, float* __restrict__ out) {
    if (threadIdx.x == 0 && blockIdx.x == 0) {
        float N = 0.0f, ll = 0.0f, lc = 0.0f;
        for (int b = 0; b < BB; ++b) { N += (float)numpos[b]; ll += locp[b]; lc += negce[b]; }
        for (int i = 0; i < nCepos; ++i) lc += cepos[i];
        out[0] = ll / N;
        out[1] = lc / N;
    }
}

extern "C" void kernel_launch(void* const* d_in, const int* in_sizes, int n_in,
                              void* d_out, int out_size, void* d_ws, size_t ws_size,
                              hipStream_t stream) {
    (void)in_sizes; (void)n_in; (void)out_size; (void)ws_size;
    const float* pred_conf = (const float*)d_in[0];
    const float* pred_loc  = (const float*)d_in[1];
    const float* priors    = (const float*)d_in[2];
    const float* truth     = (const float*)d_in[3];
    float* out = (float*)d_out;

    char* w = (char*)d_ws;
    size_t off = 0;
    auto carve = [&](size_t bytes) -> void* {
        void* p = (void*)(w + off);
        off += (bytes + 255) & ~(size_t)255;
        return p;
    };
    unsigned long long* bestKey = (unsigned long long*)carve((size_t)BB * GG * 8);
    float* ovce  = (float*)carve((size_t)BB * PP * 4);   // K1: overlap, K3: ce (reuse)
    int*   giconf= (int*)  carve((size_t)BB * PP * 4);   // K1: gt idx, K2: conf_t (reuse)
    int*   posm  = (int*)  carve((size_t)BB * PP * 4);
    float* lmarr = (float*)carve((size_t)BB * PP * 4);
    float* locp  = (float*)carve((size_t)BB * 4);
    int*   numpos= (int*)  carve((size_t)BB * 4);
    float* negce = (float*)carve((size_t)BB * 4);
    int nblkx = (PP + 127) / 128;                        // 69
    float* cepos = (float*)carve((size_t)nblkx * BB * 4);

    k_init<<<dim3((BB * GG + 255) / 256), 256, 0, stream>>>(bestKey);
    k_match<<<dim3((PP + 255) / 256, BB), 256, 0, stream>>>(priors, truth, bestKey, ovce, giconf);
    k_finalize_match<<<dim3(BB), 256, 0, stream>>>(priors, truth, pred_loc, bestKey,
                                                   ovce, giconf, posm, locp, numpos);
    k_ce_wmma<<<dim3(nblkx, BB), 256, 0, stream>>>(pred_conf, giconf, posm, ovce, lmarr, cepos);
    k_negmine<<<dim3(BB), 256, 0, stream>>>(lmarr, ovce, posm, numpos, negce);
    k_final<<<dim3(1), 32, 0, stream>>>(locp, numpos, negce, cepos, nblkx * BB, out);
}
